// deform_conv_37409165148167
// MI455X (gfx1250) — matrix operand order repaired and checked by reading, not compiled, not run on hardware
//
#include <hip/hip_runtime.h>
#include <math.h>

// Problem constants (match reference)
#define BB   2
#define CC   64
#define HH   256
#define WW   256
#define OO   64
#define K2   9
#define PH   258
#define PW   258
#define CK   576          // C * K^2
#define NPOS (BB*HH*WW)   // 131072 output positions

typedef __bf16 bf16_t;
typedef __attribute__((ext_vector_type(4)))  __bf16 v4bf;
typedef __attribute__((ext_vector_type(8)))  __bf16 v8bf;
typedef __attribute__((ext_vector_type(16))) __bf16 v16bf;
typedef __attribute__((ext_vector_type(4)))  float  v4f;
typedef __attribute__((ext_vector_type(8)))  float  v8f;

static __device__ __forceinline__ v16bf cat16(v8bf lo, v8bf hi) {
    return __builtin_shufflevector(lo, hi, 0,1,2,3,4,5,6,7,8,9,10,11,12,13,14,15);
}

// ---------------------------------------------------------------------------
// Kernel 0: pad (+1 halo) and transpose NCHW -> NHWC into workspace.
// ---------------------------------------------------------------------------
__global__ __launch_bounds__(256)
void dcn_pad_nhwc(const float* __restrict__ x, float* __restrict__ xpad) {
    __shared__ float tile[64 * 65];
    int bid = blockIdx.x;
    int xt  = bid % 5;
    int y   = (bid / 5) % PH;
    int b   = bid / (5 * PH);

    for (int i = threadIdx.x; i < 64 * 64; i += 256) {
        int c  = i >> 6, xx = i & 63;
        int gx = xt * 64 + xx;
        int iy = y - 1, ix = gx - 1;
        float v = 0.f;
        if (iy >= 0 && iy < HH && ix >= 0 && ix < WW)
            v = x[((b * CC + c) * HH + iy) * WW + ix];
        tile[c * 65 + xx] = v;
    }
    __syncthreads();
    for (int i = threadIdx.x; i < 64 * 64; i += 256) {
        int xx = i >> 6, c = i & 63;
        int gx = xt * 64 + xx;
        if (gx < PW)
            xpad[((b * PH + y) * PW + gx) * CC + c] = tile[c * 65 + xx];
    }
}

// ---------------------------------------------------------------------------
// Kernel 0b: repack weights to bf16 with contraction index ck = k*64 + c.
// B1[32][576]: rows 0..17 offset_w, 18..26 amp_w, 27..31 zero.
// B2[64][576]: w3d[o][c][k] -> B2[o][k*64+c].
// ---------------------------------------------------------------------------
__global__ __launch_bounds__(256)
void dcn_prep_weights(const float* __restrict__ ow, const float* __restrict__ aw,
                      const float* __restrict__ w3d,
                      bf16_t* __restrict__ B1, bf16_t* __restrict__ B2) {
    int i = blockIdx.x * 256 + threadIdx.x;
    if (i < 32 * CK) {
        int n = i / CK, r = i % CK;
        int k = r >> 6, c = r & 63;
        float v = 0.f;
        if (n < 18)       v = ow[n * CK + c * 9 + k];
        else if (n < 27)  v = aw[(n - 18) * CK + c * 9 + k];
        B1[i] = (bf16_t)v;
    } else if (i < 32 * CK + 64 * CK) {
        int j = i - 32 * CK;
        int o = j / CK, r = j % CK;
        int k = r >> 6, c = r & 63;
        B2[j] = (bf16_t)w3d[o * CK + c * 9 + k];
    }
}

// ---------------------------------------------------------------------------
// Kernel 1: offset+modulation convs as an implicit GEMM.
//   M-tile = 32 positions, N = 32 (27 used), K = 576. 2 waves, each owning one
//   N-subtile and BOTH M-subtiles (B fragment reused across 2 WMMAs/kb).
// ---------------------------------------------------------------------------
__global__ __launch_bounds__(64)
void dcn_offamp_gemm(const float* __restrict__ xpad, const bf16_t* __restrict__ B1,
                     const float* __restrict__ offb, const float* __restrict__ ampb,
                     float* __restrict__ offamp) {
    __shared__ bf16_t A1[32 * CK];       // 36 KB
    __shared__ float  D1[32 * 32];       // 4 KB

    int posBase = blockIdx.x * 32;
    int b   = posBase >> 16;
    int rem = posBase & 65535;
    int h   = rem >> 8;
    int w0  = rem & 255;

    // im2col tile: A1[m][k*64+c] = xpad[b][h+ky][w0+m+kx][c]
    for (int t = threadIdx.x; t < 288; t += 64) {
        int m = t & 31, k = t >> 5;
        int ky = k / 3, kx = k % 3;
        const float* src = xpad + (((b * PH) + h + ky) * PW + (w0 + m + kx)) * CC;
        bf16_t* dst = A1 + m * CK + k * 64;
        #pragma unroll
        for (int c = 0; c < 64; c += 4) {
            v4f f = *(const v4f*)(src + c);
            v4bf p = { (bf16_t)f[0], (bf16_t)f[1], (bf16_t)f[2], (bf16_t)f[3] };
            *(v4bf*)(dst + c) = p;
        }
    }
    __syncthreads();

    int lane = threadIdx.x & 31;
    int wv   = threadIdx.x >> 5;         // N-subtile 0 or 1
    int half = lane >> 4;                // K-half select per ISA layout
    int mrow = lane & 15;

    v8f acc0 = {}, acc1 = {};
    const bf16_t* brow = B1 + (wv * 16 + mrow) * CK;
    #pragma unroll
    for (int kb = 0; kb < 18; ++kb) {
        const bf16_t* bp = brow + kb * 32 + half * 8;
        v16bf bm = cat16(*(const v8bf*)(bp), *(const v8bf*)(bp + 16));

        int a0 = mrow * CK + kb * 32 + half * 8;
        v16bf a = cat16(*(const v8bf*)(A1 + a0), *(const v8bf*)(A1 + a0 + 16));
        acc0 = __builtin_amdgcn_wmma_f32_16x16x32_bf16(false, a, false, bm,
                                                       (short)0, acc0, false, false);
        int a1 = (16 + mrow) * CK + kb * 32 + half * 8;
        v16bf a2 = cat16(*(const v8bf*)(A1 + a1), *(const v8bf*)(A1 + a1 + 16));
        acc1 = __builtin_amdgcn_wmma_f32_16x16x32_bf16(false, a2, false, bm,
                                                       (short)0, acc1, false, false);
    }
    #pragma unroll
    for (int r = 0; r < 8; ++r) {
        D1[(r + half * 8) * 32 + wv * 16 + mrow]      = acc0[r];
        D1[(16 + r + half * 8) * 32 + wv * 16 + mrow] = acc1[r];
    }
    __syncthreads();

    for (int i = threadIdx.x; i < 1024; i += 64) {
        int m = i >> 5, n = i & 31;
        float v = D1[i];
        int pos = posBase + m;
        float o;
        if (n < 18)      o = v + offb[n];
        else if (n < 27) { float z = v + ampb[n - 18]; o = 1.f / (1.f + __expf(-z)); }
        else             o = 0.f;
        offamp[pos * 32 + n] = o;
    }
}

// ---------------------------------------------------------------------------
// Kernel 2: deformable gather + modulation + main GEMM.
//   Per block: 32 positions x all 64 output channels. 4 waves, one N-subtile
//   each; every wave computes BOTH 16-row M-subtiles so the B fragment loaded
//   per K-step feeds two WMMAs.
// ---------------------------------------------------------------------------
__global__ __launch_bounds__(128)
void dcn_main_gemm(const float* __restrict__ xpad, const bf16_t* __restrict__ B2,
                   const float* __restrict__ offamp, float* __restrict__ out) {
    __shared__ bf16_t A2[32 * CK];       // 36 KB
    __shared__ float  Dbuf[64 * 32];     // 8 KB

    int posBase = blockIdx.x * 32;
    int b   = posBase >> 16;
    int rem = posBase & 65535;
    int h   = rem >> 8;
    int w0  = rem & 255;

    // Warm WGP$ with the (shared, 72KB) B matrix while phase 1 runs.
    {
        const char* pb = (const char*)B2;
        size_t off = (size_t)threadIdx.x * 576;
        __builtin_prefetch(pb + off, 0, 3);
        __builtin_prefetch(pb + off + 256, 0, 3);
        __builtin_prefetch(pb + off + 512, 0, 3);
    }

    // Phase 1: deformed, modulated im2col tile.
    for (int t = threadIdx.x; t < 288; t += 128) {
        int m = t & 31, k = t >> 5;
        int ky = k / 3, kx = k % 3;
        int pos = posBase + m;
        const float* oa = offamp + pos * 32;
        float yo = oa[2 * k], xo = oa[2 * k + 1], ampv = oa[18 + k];

        float yf = (float)(h + ky) + yo;
        float xf = (float)(w0 + m + kx) + xo;
        yf = fminf(fmaxf(yf, 0.f), 257.f);
        xf = fminf(fmaxf(xf, 0.f), 257.f);
        float y0f = floorf(yf), x0f = floorf(xf);
        int iy0 = (int)y0f, ix0 = (int)x0f;
        int iy1 = min(iy0 + 1, 257), ix1 = min(ix0 + 1, 257);
        float wy1 = yf - y0f, wy0 = (float)iy1 - yf;
        float wx1 = xf - x0f, wx0 = (float)ix1 - xf;
        float w00 = wy0 * wx0 * ampv, w01 = wy0 * wx1 * ampv;
        float w10 = wy1 * wx0 * ampv, w11 = wy1 * wx1 * ampv;

        const v4f* r00 = (const v4f*)(xpad + ((b * PH + iy0) * PW + ix0) * CC);
        const v4f* r01 = (const v4f*)(xpad + ((b * PH + iy0) * PW + ix1) * CC);
        const v4f* r10 = (const v4f*)(xpad + ((b * PH + iy1) * PW + ix0) * CC);
        const v4f* r11 = (const v4f*)(xpad + ((b * PH + iy1) * PW + ix1) * CC);
        bf16_t* dst = A2 + m * CK + k * 64;
        #pragma unroll
        for (int j = 0; j < 16; ++j) {
            v4f a0 = r00[j], a1 = r01[j], a2 = r10[j], a3 = r11[j];
            v4bf p;
            #pragma unroll
            for (int e = 0; e < 4; ++e) {
                float v = w00 * a0[e] + w01 * a1[e] + w10 * a2[e] + w11 * a3[e];
                p[e] = (bf16_t)v;
            }
            *(v4bf*)(dst + j * 4) = p;
        }
    }
    __syncthreads();

    // Phase 2: each wave computes two 16x16 tiles of out[pos, o].
    int lane = threadIdx.x & 31;
    int nt   = threadIdx.x >> 5;
    int half = lane >> 4;
    int mrow = lane & 15;

    v8f acc0 = {}, acc1 = {};
    const bf16_t* brow = B2 + (nt * 16 + mrow) * CK;
    #pragma unroll
    for (int kb = 0; kb < 18; ++kb) {
        const bf16_t* bp = brow + kb * 32 + half * 8;
        v16bf bm = cat16(*(const v8bf*)(bp), *(const v8bf*)(bp + 16));

        int a0 = mrow * CK + kb * 32 + half * 8;
        v16bf a = cat16(*(const v8bf*)(A2 + a0), *(const v8bf*)(A2 + a0 + 16));
        acc0 = __builtin_amdgcn_wmma_f32_16x16x32_bf16(false, a, false, bm,
                                                       (short)0, acc0, false, false);
        int a1 = (16 + mrow) * CK + kb * 32 + half * 8;
        v16bf a2v = cat16(*(const v8bf*)(A2 + a1), *(const v8bf*)(A2 + a1 + 16));
        acc1 = __builtin_amdgcn_wmma_f32_16x16x32_bf16(false, a2v, false, bm,
                                                       (short)0, acc1, false, false);
    }
    #pragma unroll
    for (int r = 0; r < 8; ++r) {
        Dbuf[(nt * 16 + mrow) * 32 + (r + half * 8)]      = acc0[r];
        Dbuf[(nt * 16 + mrow) * 32 + 16 + (r + half * 8)] = acc1[r];
    }
    __syncthreads();

    // Phase 3: ReLU + coalesced store (32 consecutive w per 32 lanes).
    for (int i = threadIdx.x; i < 2048; i += 128) {
        int o = i >> 5, m = i & 31;
        out[((b * OO + o) * HH + h) * WW + w0 + m] = fmaxf(Dbuf[o * 32 + m], 0.f);
    }
}

// ---------------------------------------------------------------------------
extern "C" void kernel_launch(void* const* d_in, const int* in_sizes, int n_in,
                              void* d_out, int out_size, void* d_ws, size_t ws_size,
                              hipStream_t stream) {
    const float* x    = (const float*)d_in[0];
    const float* ow   = (const float*)d_in[1];
    const float* ob   = (const float*)d_in[2];
    const float* aw   = (const float*)d_in[3];
    const float* ab   = (const float*)d_in[4];
    const float* w3d  = (const float*)d_in[5];
    float* out = (float*)d_out;

    // Workspace layout (all 16B-aligned):
    //   xpad   : BB*PH*PW*CC f32          (34.1 MB)
    //   B1     : 32*CK bf16               (36.9 KB)
    //   B2     : 64*CK bf16               (73.7 KB)
    //   offamp : NPOS*32 f32              (16.8 MB)
    char* ws = (char*)d_ws;
    size_t xpadBytes = (size_t)BB * PH * PW * CC * sizeof(float);
    float*  xpad   = (float*)ws;
    bf16_t* B1     = (bf16_t*)(ws + xpadBytes);
    bf16_t* B2     = B1 + 32 * CK;
    float*  offamp = (float*)(ws + xpadBytes + (size_t)96 * CK * sizeof(bf16_t));

    dcn_pad_nhwc    <<<BB * PH * 5, 256, 0, stream>>>(x, xpad);
    dcn_prep_weights<<<(96 * CK + 255) / 256, 256, 0, stream>>>(ow, aw, w3d, B1, B2);
    dcn_offamp_gemm <<<NPOS / 32, 64, 0, stream>>>(xpad, B1, ob, ab, offamp);
    dcn_main_gemm   <<<NPOS / 32, 128, 0, stream>>>(xpad, B2, offamp, out);
}
